// MultilayeredNetwork_59725815218233
// MI455X (gfx1250) — matrix operand order repaired and checked by reading, not compiled, not run on hardware
//
#include <hip/hip_runtime.h>
#include <math.h>

// Problem constants (match the reference)
#define NN 16384        // total neurons
#define SS 2048         // sensory neurons
#define TT 16           // timesteps
#define KSPLIT 8        // K-dimension split for parallelism
#define THRESH 0.01f
#define STEEP 5.0f

typedef _Float16 v16h __attribute__((ext_vector_type(16)));
typedef _Float16 h8   __attribute__((ext_vector_type(8)));
typedef float    v8f  __attribute__((ext_vector_type(8)));

// ---------------------------------------------------------------------------
// One-time f32 -> f16 weight conversion (1 GiB read, 0.5 GiB write)
// ---------------------------------------------------------------------------
__global__ __launch_bounds__(256) void convert_w(const float* __restrict__ W,
                                                 _Float16* __restrict__ Wh) {
    size_t i = ((size_t)blockIdx.x * 256 + threadIdx.x) * 8;
    float4 a = *reinterpret_cast<const float4*>(W + i);
    float4 b = *reinterpret_cast<const float4*>(W + i + 4);
    h8 o;
    o[0] = (_Float16)a.x; o[1] = (_Float16)a.y;
    o[2] = (_Float16)a.z; o[3] = (_Float16)a.w;
    o[4] = (_Float16)b.x; o[5] = (_Float16)b.y;
    o[6] = (_Float16)b.z; o[7] = (_Float16)b.w;
    *reinterpret_cast<h8*>(Wh + i) = o;
}

// ---------------------------------------------------------------------------
// Input preprocessing: u = tanh(where(in >= thr, in, 0)); x0 state = u[:,0]
// inthroughtime is [S, T] row-major.
// ---------------------------------------------------------------------------
__global__ __launch_bounds__(256) void prep_inputs(const float* __restrict__ in,
                                                   float* __restrict__ u,
                                                   _Float16* __restrict__ xh) {
    int i = blockIdx.x * 256 + threadIdx.x;   // i in [0, S*T)
    float v = in[i];
    float r = tanhf((v >= THRESH) ? v : 0.0f);
    u[i] = r;
    if ((i % TT) == 0) xh[i / TT] = (_Float16)r;
}

// ---------------------------------------------------------------------------
// Operand loaders (A: 16x32 f16 tile in ISA VGPR layout; B: x replicated)
// ---------------------------------------------------------------------------
template <bool W_IS_F16>
__device__ __forceinline__ v16h load_a(const void* __restrict__ Wp, size_t idx) {
    if constexpr (W_IS_F16) {
        const _Float16* wp = (const _Float16*)Wp + idx;
        h8 lo  = *reinterpret_cast<const h8*>(wp);        // K: ka .. ka+7
        h8 hi8 = *reinterpret_cast<const h8*>(wp + 16);   // K: ka+16 .. ka+23
        return __builtin_shufflevector(lo, hi8, 0, 1, 2, 3, 4, 5, 6, 7,
                                                8, 9, 10, 11, 12, 13, 14, 15);
    } else {
        const float* wp = (const float*)Wp + idx;
        float4 f0 = *reinterpret_cast<const float4*>(wp);
        float4 f1 = *reinterpret_cast<const float4*>(wp + 4);
        float4 f2 = *reinterpret_cast<const float4*>(wp + 16);
        float4 f3 = *reinterpret_cast<const float4*>(wp + 20);
        v16h a;
        a[0]  = (_Float16)f0.x; a[1]  = (_Float16)f0.y;
        a[2]  = (_Float16)f0.z; a[3]  = (_Float16)f0.w;
        a[4]  = (_Float16)f1.x; a[5]  = (_Float16)f1.y;
        a[6]  = (_Float16)f1.z; a[7]  = (_Float16)f1.w;
        a[8]  = (_Float16)f2.x; a[9]  = (_Float16)f2.y;
        a[10] = (_Float16)f2.z; a[11] = (_Float16)f2.w;
        a[12] = (_Float16)f3.x; a[13] = (_Float16)f3.y;
        a[14] = (_Float16)f3.z; a[15] = (_Float16)f3.w;
        return a;
    }
}

__device__ __forceinline__ v16h load_b(const _Float16* __restrict__ xp) {
    h8 b0 = *reinterpret_cast<const h8*>(xp);
    h8 b1 = *reinterpret_cast<const h8*>(xp + 16);
    return __builtin_shufflevector(b0, b1, 0, 1, 2, 3, 4, 5, 6, 7,
                                           8, 9, 10, 11, 12, 13, 14, 15);
}

// ---------------------------------------------------------------------------
// GEMV tile kernel: y_partial = W[tile rows, k-chunk] @ x  via WMMA f16.
// Each wave owns one 16-row tile and one of KSPLIT K-chunks (KLEN compile-
// time: 256 for the t=0 sensory pass, 2048 for the full matvec).
// B operand: x replicated into all 16 columns (broadcast load); every output
// column then equals y, and lanes 0/16 hold M=0..7 / M=8..15 per C layout.
// Manual 4x unroll: 8 A-loads batched ahead of 4 chained WMMAs per iteration
// (memory-level parallelism to cover HBM latency at 23.3 TB/s).
// ---------------------------------------------------------------------------
template <bool W_IS_F16, int KLEN>
__global__ __launch_bounds__(256) void gemv_wmma(const void* __restrict__ Wp,
                                                 const _Float16* __restrict__ xh,
                                                 float* __restrict__ partial) {
    const int wave  = (blockIdx.x << 3) | (threadIdx.x >> 5);
    const int lane  = threadIdx.x & 31;
    const int tile  = wave >> 3;            // wave / KSPLIT
    const int chunk = wave & (KSPLIT - 1);
    const int row0  = tile << 4;
    const int k0    = chunk * KLEN;

    const int m  = lane & 15;               // row within the 16-row tile
    const int hi = lane >> 4;               // lane half (A/B layout split)
    const int ka = hi * 8;                  // A K-offset:  lanes>=16 start at K=8
    const int kb = hi * 8;                  // B K-offset:  lanes>=16 start at K=8
    const size_t abase = (size_t)(row0 + m) * NN + ka;

    v8f c = {};
#pragma unroll 1
    for (int k = k0; k < k0 + KLEN; k += 128) {
        // Batch all loads for 4 WMMA steps (scheduler can clause these).
        v16h a0 = load_a<W_IS_F16>(Wp, abase + k);
        v16h a1 = load_a<W_IS_F16>(Wp, abase + k + 32);
        v16h a2 = load_a<W_IS_F16>(Wp, abase + k + 64);
        v16h a3 = load_a<W_IS_F16>(Wp, abase + k + 96);
        v16h b0 = load_b(xh + k + kb);
        v16h b1 = load_b(xh + k + kb + 32);
        v16h b2 = load_b(xh + k + kb + 64);
        v16h b3 = load_b(xh + k + kb + 96);
        // Stream-prefetch W one unrolled iteration ahead (global_prefetch_b8).
        if constexpr (W_IS_F16) {
            __builtin_prefetch((const _Float16*)Wp + abase + k + 128, 0, 0);
        } else {
            __builtin_prefetch((const float*)Wp + abase + k + 128, 0, 0);
        }
        // D = A*B + C   (v_wmma_f32_16x16x32_f16 x4)
        c = __builtin_amdgcn_wmma_f32_16x16x32_f16(false, a0, false, b0, (short)0, c, false, false);
        c = __builtin_amdgcn_wmma_f32_16x16x32_f16(false, a1, false, b1, (short)0, c, false, false);
        c = __builtin_amdgcn_wmma_f32_16x16x32_f16(false, a2, false, b2, (short)0, c, false, false);
        c = __builtin_amdgcn_wmma_f32_16x16x32_f16(false, a3, false, b3, (short)0, c, false, false);
    }

    // Column N=0 of C: lane 0 has M=0..7 in c[0..7], lane 16 has M=8..15.
    if ((lane & 15) == 0) {
        float* dst = partial + (size_t)chunk * NN + row0 + hi * 8;
        *reinterpret_cast<float4*>(dst)     = make_float4(c[0], c[1], c[2], c[3]);
        *reinterpret_cast<float4*>(dst + 4) = make_float4(c[4], c[5], c[6], c[7]);
    }
}

// ---------------------------------------------------------------------------
// Sum KSPLIT partials, add sensory input (t>=1, n<S), activate, emit state.
// out is [N, T] row-major.
// ---------------------------------------------------------------------------
__global__ __launch_bounds__(256) void reduce_act(const float* __restrict__ partial,
                                                  const float* __restrict__ u,
                                                  float* __restrict__ out,
                                                  _Float16* __restrict__ xh,
                                                  int t, int add_sens) {
    int n = blockIdx.x * 256 + threadIdx.x;
    float s = 0.0f;
#pragma unroll
    for (int c = 0; c < KSPLIT; ++c) s += partial[(size_t)c * NN + n];
    if (add_sens && n < SS) s += u[n * TT + t];
    float z = (s >= THRESH) ? s : 0.0f;
    float a = tanhf(STEEP * z);
    out[(size_t)n * TT + t] = a;
    xh[n] = (_Float16)a;
}

// ---------------------------------------------------------------------------
// Host launcher
// ---------------------------------------------------------------------------
extern "C" void kernel_launch(void* const* d_in, const int* in_sizes, int n_in,
                              void* d_out, int out_size, void* d_ws, size_t ws_size,
                              hipStream_t stream) {
    (void)in_sizes; (void)n_in; (void)out_size;
    const float* W   = (const float*)d_in[0];   // [N, N] f32
    const float* inr = (const float*)d_in[1];   // [S, T] f32
    // d_in[2] = sensory_indices (arange(S)) -> rows/cols 0..S-1, used implicitly
    float* out = (float*)d_out;                 // [N, T] f32
    char*  ws  = (char*)d_ws;

    const size_t whBytes = (size_t)NN * NN * sizeof(_Float16);   // 512 MiB
    const size_t uBytes  = (size_t)SS * TT * sizeof(float);
    const size_t xBytes  = (size_t)NN * sizeof(_Float16);
    const size_t pBytes  = (size_t)NN * KSPLIT * sizeof(float);
    const bool useWh = ws_size >= whBytes + uBytes + xBytes + pBytes;

    size_t off = useWh ? whBytes : 0;
    _Float16* Wh      = (_Float16*)ws;
    float*    u       = (float*)(ws + off);      off += uBytes;
    _Float16* xh      = (_Float16*)(ws + off);   off += xBytes;
    float*    partial = (float*)(ws + off);

    if (useWh) {
        // N*N elements, 8 per thread: 131072 blocks x 256 threads
        convert_w<<<dim3(131072), dim3(256), 0, stream>>>(W, Wh);
    }
    prep_inputs<<<dim3((SS * TT) / 256), dim3(256), 0, stream>>>(inr, u, xh);

    const int blocksG = (NN / 16) * KSPLIT / 8;   // 8192 waves / 8 per WG = 1024

    // t = 0: only sensory columns (leading S columns) drive the network; no add.
    if (useWh) gemv_wmma<true,  SS / KSPLIT><<<dim3(blocksG), dim3(256), 0, stream>>>(Wh, xh, partial);
    else       gemv_wmma<false, SS / KSPLIT><<<dim3(blocksG), dim3(256), 0, stream>>>(W,  xh, partial);
    reduce_act<<<dim3(NN / 256), dim3(256), 0, stream>>>(partial, u, out, xh, 0, 0);

    // t = 1..T-1: full connectome matvec + sensory scatter-add.
    for (int t = 1; t < TT; ++t) {
        if (useWh) gemv_wmma<true,  NN / KSPLIT><<<dim3(blocksG), dim3(256), 0, stream>>>(Wh, xh, partial);
        else       gemv_wmma<false, NN / KSPLIT><<<dim3(blocksG), dim3(256), 0, stream>>>(W,  xh, partial);
        reduce_act<<<dim3(NN / 256), dim3(256), 0, stream>>>(partial, u, out, xh, t, 1);
    }
}
